// AttentionLayer_18872086299291
// MI455X (gfx1250) — matrix operand order repaired and checked by reading, not compiled, not run on hardware
//
#include <hip/hip_runtime.h>

#define B_  8
#define S_  1024
#define D_  1024
#define NH_ 16
#define H_  64
#define BS_ (B_*S_)   // 8192

typedef __attribute__((ext_vector_type(16))) __bf16 bf16x16;
typedef __attribute__((ext_vector_type(8)))  float  floatx8;

union BFrag { bf16x16 v; uint4 u[2]; };

// A-matrix 16x32 bf16 fragment (wave32): lanes 0-15 row m, K {0..7,16..23};
// lanes 16-31 row m, K {8..15,24..31}.  Memory row-major, ld elements.
__device__ __forceinline__ bf16x16 load_a_frag(const __bf16* base, int ld,
                                               int rowBase, int kBase, int lane) {
  const int m     = lane & 15;
  const int khalf = (lane >> 4) << 3;
  const __bf16* p = base + (size_t)(rowBase + m) * ld + kBase;
  BFrag f;
  f.u[0] = *reinterpret_cast<const uint4*>(p + khalf);
  f.u[1] = *reinterpret_cast<const uint4*>(p + 16 + khalf);
  return f.v;
}

// B-matrix 32x16 bf16 fragment (wave32): lanes 0-15 col n, K {0..7,16..23};
// lanes 16-31 col n, K {8..15,24..31}.  Memory layout: column n has K contiguous (ld elems).
__device__ __forceinline__ bf16x16 load_b_frag(const __bf16* base, int ld,
                                               int colBase, int kBase, int lane) {
  const int n  = lane & 15;
  const int ks = (lane >> 4) << 3;
  const __bf16* p = base + (size_t)(colBase + n) * ld + kBase + ks;
  BFrag f;
  f.u[0] = *reinterpret_cast<const uint4*>(p);
  f.u[1] = *reinterpret_cast<const uint4*>(p + 16);
  return f.v;
}

__device__ __forceinline__ floatx8 wmma_bf16(bf16x16 a, bf16x16 b, floatx8 c) {
  return __builtin_amdgcn_wmma_f32_16x16x32_bf16(false, a, false, b, (short)0, c,
                                                 false, false);
}

// ---------------- staging: fp32 -> bf16 ----------------
__global__ void cvt_bf16_kernel(const float* __restrict__ in,
                                __bf16* __restrict__ out, int n) {
  int i = blockIdx.x * blockDim.x + threadIdx.x;
  if (i < n) out[i] = (__bf16)in[i];
}

// weight [NH][D][H] -> wT [NH][H][D] bf16 (K-contiguous for B-frags)
__global__ void wtr_kernel(const float* __restrict__ w, __bf16* __restrict__ wT) {
  int i = blockIdx.x * blockDim.x + threadIdx.x;   // 0 .. NH*D*H-1
  int n = i >> 16;                                  // D*H = 65536
  int rem = i & 65535;
  int d = rem >> 6;
  int h = rem & 63;
  wT[((size_t)n * H_ + h) * D_ + d] = (__bf16)w[i];
}

// ---------------- per-head projection: logits = X @ W[n] ----------------
// grid = 128 mblocks * 16 heads, block = 128 (4 waves).
// Each wave: 16 rows x full H=64 -> 4 accumulators sharing one A-frag per k-step.
__global__ void proj_kernel(const __bf16* __restrict__ xb,
                            const __bf16* __restrict__ wT,
                            __bf16* __restrict__ lg,   // [NH*B][S][H]
                            __bf16* __restrict__ lgT)  // [NH*B][H][S]
{
  const int bid    = blockIdx.x;
  const int head   = bid >> 7;          // 128 mblocks per head
  const int mblock = bid & 127;
  const int tid    = threadIdx.x;
  const int wave   = tid >> 5;
  const int lane   = tid & 31;
  const int rowBase = mblock * 64 + wave * 16;
  const __bf16* wbase = wT + (size_t)head * H_ * D_;

  floatx8 acc[4] = {};
  for (int kt = 0; kt < D_ / 32; ++kt) {
    const int kBase = kt * 32;
    bf16x16 a = load_a_frag(xb, D_, rowBase, kBase, lane);
    for (int h = 0; h < 4; ++h) {
      bf16x16 bm = load_b_frag(wbase, D_, h * 16, kBase, lane);
      acc[h] = wmma_bf16(a, bm, acc[h]);
    }
  }
  const int nn = lane & 15;
  const int mh = (lane >> 4) << 3;
  for (int h = 0; h < 4; ++h) {
    for (int r = 0; r < 8; ++r) {
      int m = r + mh;
      int bsRow = rowBase + m;
      int b = bsRow >> 10;
      int s = bsRow & 1023;
      int hh = h * 16 + nn;
      __bf16 v = (__bf16)acc[h][r];
      size_t nb = (size_t)(head * B_ + b);
      lg [(nb * S_ + s) * H_ + hh] = v;
      lgT[(nb * H_ + hh) * S_ + s] = v;
    }
  }
}

// ---------------- attention: scores -> softmax -> P@V ----------------
// one workgroup per (head, batch, 16-query block); 128 threads = 4 waves
// dynamic LDS: scores 16x1024 f32 (64KB) + probs 16x1024 bf16 (32KB) + reductions
__global__ void attn_kernel(const __bf16* __restrict__ lg,
                            const __bf16* __restrict__ lgT,
                            float* __restrict__ attn_out,   // [NH*B][S][S]
                            __bf16* __restrict__ ho)        // [B][S][NH*H]
{
  extern __shared__ char smem[];
  float*  sS   = (float*)smem;                        // [16][1024]
  __bf16* sP   = (__bf16*)(smem + 65536);             // [16][1024]
  float*  sRed = (float*)(smem + 65536 + 32768);      // [16][8]
  float*  sMax = sRed + 128;                          // [16]
  float*  sSum = sMax + 16;                           // [16]

  const int tid  = threadIdx.x;
  const int wave = tid >> 5;
  const int lane = tid & 31;
  int bid = blockIdx.x;
  const int q = bid & 63; bid >>= 6;
  const int b = bid & 7;
  const int n = bid >> 3;
  const int qBase = q * 16;
  const size_t nb = (size_t)(n * B_ + b);
  const __bf16* lgh  = lg  + nb * S_ * H_;   // [S][H]
  const __bf16* lghT = lgT + nb * H_ * S_;   // [H][S]

  // ---- scores: Q(16xH) @ K^T -> 16x1024 in LDS ----
  // Q fragments are loop-invariant: hoist them (2 frags cover K=64).
  bf16x16 qa0 = load_a_frag(lgh, H_, qBase, 0,  lane);
  bf16x16 qa1 = load_a_frag(lgh, H_, qBase, 32, lane);
  for (int tt = wave; tt < S_ / 16; tt += 4) {
    floatx8 acc = {};
    bf16x16 b0 = load_b_frag(lgh, H_, tt * 16, 0,  lane);  // K^T: col t, k=h contig
    bf16x16 b1 = load_b_frag(lgh, H_, tt * 16, 32, lane);
    acc = wmma_bf16(qa0, b0, acc);
    acc = wmma_bf16(qa1, b1, acc);
    const int nn = lane & 15;
    const int mh = (lane >> 4) << 3;
    for (int r = 0; r < 8; ++r) sS[(r + mh) * 1024 + tt * 16 + nn] = acc[r];
  }
  __syncthreads();

  // ---- softmax over each row of 1024 (scale = 1/sqrt(H)) ----
  const float scale = 0.125f;
  const int row = tid & 15;
  const int chunk = tid >> 4;                // 8 chunks of 128
  const int c0 = chunk * 128;

  float mx = -3.4e38f;
  for (int c = c0; c < c0 + 128; ++c) mx = fmaxf(mx, sS[row * 1024 + c]);
  sRed[row * 8 + chunk] = mx;
  __syncthreads();
  if (tid < 16) {
    float mm = -3.4e38f;
    for (int j = 0; j < 8; ++j) mm = fmaxf(mm, sRed[tid * 8 + j]);
    sMax[tid] = mm;
  }
  __syncthreads();
  const float rmax = sMax[row];
  float sum = 0.f;
  for (int c = c0; c < c0 + 128; ++c) {
    float e = __expf(scale * (sS[row * 1024 + c] - rmax));
    sS[row * 1024 + c] = e;
    sum += e;
  }
  sRed[row * 8 + chunk] = sum;
  __syncthreads();
  if (tid < 16) {
    float s = 0.f;
    for (int j = 0; j < 8; ++j) s += sRed[tid * 8 + j];
    sSum[tid] = s;
  }
  __syncthreads();
  const float inv = 1.0f / sSum[row];
  float* arow = attn_out + (nb * S_ + qBase + row) * S_;
  for (int c = c0; c < c0 + 128; ++c) {
    float p = sS[row * 1024 + c] * inv;
    arow[c] = p;                    // required fp32 attn output
    sP[row * 1024 + c] = (__bf16)p; // bf16 probs for P@V
  }
  __syncthreads();

  // ---- out tile: P(16x1024) @ V(1024x64); one H-tile per wave ----
  const int hBase = wave * 16;
  floatx8 acc = {};
  for (int kt = 0; kt < S_ / 32; ++kt) {
    const int kBase = kt * 32;
    bf16x16 a  = load_a_frag(sP,   1024, 0,     kBase, lane);  // probs from LDS
    bf16x16 bm = load_b_frag(lghT, S_,   hBase, kBase, lane);  // V: col h, k=t contig
    acc = wmma_bf16(a, bm, acc);
  }
  const int nn = lane & 15;
  const int mh = (lane >> 4) << 3;
  for (int r = 0; r < 8; ++r) {
    int m = r + mh;
    size_t rowg = (size_t)b * S_ + qBase + m;
    ho[rowg * (NH_ * H_) + n * H_ + hBase + nn] = (__bf16)acc[r];
  }
}

// ---------------- output linear + bias + residual ----------------
// grid = 64 mblocks * 16 d-groups, block = 128 (4 waves).
// Each wave: 32 rows x 64 cols = 8 accumulators; per k-step 2 A + 4 B loads -> 8 wmma.
__global__ void fin_kernel(const __bf16* __restrict__ ho,
                           const __bf16* __restrict__ lwb,  // lin_w bf16 [D][K] (K contig)
                           const float* __restrict__ lin_b,
                           const float* __restrict__ x,
                           float* __restrict__ ytmp) {
  const int bid    = blockIdx.x;
  const int mblock = bid >> 4;
  const int ng     = bid & 15;
  const int tid    = threadIdx.x;
  const int wave   = tid >> 5;
  const int lane   = tid & 31;
  const int dBase  = ng * 64;
  const int rowBase = mblock * 128 + wave * 32;

  floatx8 acc[2][4] = {};
  for (int kt = 0; kt < D_ / 32; ++kt) {
    const int kBase = kt * 32;
    bf16x16 a0 = load_a_frag(ho, NH_ * H_, rowBase,      kBase, lane);
    bf16x16 a1 = load_a_frag(ho, NH_ * H_, rowBase + 16, kBase, lane);
    for (int nj = 0; nj < 4; ++nj) {
      bf16x16 bm = load_b_frag(lwb, NH_ * H_, dBase + nj * 16, kBase, lane);
      acc[0][nj] = wmma_bf16(a0, bm, acc[0][nj]);
      acc[1][nj] = wmma_bf16(a1, bm, acc[1][nj]);
    }
  }
  const int nn = lane & 15;
  const int mh = (lane >> 4) << 3;
  for (int mi = 0; mi < 2; ++mi) {
    for (int nj = 0; nj < 4; ++nj) {
      for (int r = 0; r < 8; ++r) {
        size_t row = (size_t)rowBase + mi * 16 + r + mh;
        int d = dBase + nj * 16 + nn;
        ytmp[row * D_ + d] = acc[mi][nj][r] + lin_b[d] + x[row * D_ + d];
      }
    }
  }
}

// ---------------- layernorm ----------------
__global__ void ln_kernel(const float* __restrict__ ytmp,
                          const float* __restrict__ g,
                          const float* __restrict__ bta,
                          float* __restrict__ yout) {
  __shared__ float red[256], red2[256];
  const int row = blockIdx.x;
  const int tid = threadIdx.x;
  const float* p = ytmp + (size_t)row * D_;
  float v[4], s = 0.f, s2 = 0.f;
  for (int j = 0; j < 4; ++j) {
    v[j] = p[tid + j * 256];
    s += v[j];
    s2 += v[j] * v[j];
  }
  red[tid] = s; red2[tid] = s2;
  __syncthreads();
  for (int off = 128; off > 0; off >>= 1) {
    if (tid < off) { red[tid] += red[tid + off]; red2[tid] += red2[tid + off]; }
    __syncthreads();
  }
  const float mean = red[0] * (1.0f / 1024.0f);
  const float var  = red2[0] * (1.0f / 1024.0f) - mean * mean;
  const float rs   = rsqrtf(var + 1e-5f);
  for (int j = 0; j < 4; ++j) {
    int c = tid + j * 256;
    yout[(size_t)row * D_ + c] = (v[j] - mean) * rs * g[c] + bta[c];
  }
}

extern "C" void kernel_launch(void* const* d_in, const int* in_sizes, int n_in,
                              void* d_out, int out_size, void* d_ws, size_t ws_size,
                              hipStream_t stream) {
  const float* x      = (const float*)d_in[0];  // [B,S,D]
  const float* weight = (const float*)d_in[1];  // [NH,D,H]
  const float* lin_w  = (const float*)d_in[2];  // [D, NH*H]
  const float* lin_b  = (const float*)d_in[3];  // [D]
  const float* ln_g   = (const float*)d_in[4];  // [D]
  const float* ln_b   = (const float*)d_in[5];  // [D]

  float* y_out    = (float*)d_out;                          // [B,S,D]
  float* attn_out = y_out + (size_t)BS_ * D_;               // [NH*B,S,S]

  // workspace carve
  size_t off = 0;
  char* wsb = (char*)d_ws;
  auto carve = [&](size_t bytes) -> void* {
    void* p = wsb + off;
    off += (bytes + 255) & ~(size_t)255;
    return p;
  };
  __bf16* xb   = (__bf16*)carve((size_t)BS_ * D_ * 2);            // x bf16
  __bf16* wT   = (__bf16*)carve((size_t)NH_ * D_ * H_ * 2);       // weight^T bf16
  __bf16* lwb  = (__bf16*)carve((size_t)D_ * NH_ * H_ * 2);       // lin_w bf16
  __bf16* lg   = (__bf16*)carve((size_t)NH_ * B_ * S_ * H_ * 2);  // logits
  __bf16* lgT  = (__bf16*)carve((size_t)NH_ * B_ * S_ * H_ * 2);  // logits^T
  __bf16* ho   = (__bf16*)carve((size_t)BS_ * NH_ * H_ * 2);      // head concat
  float*  ytmp = (float*)carve((size_t)BS_ * D_ * 4);             // pre-LN

  const int nx = BS_ * D_;         // 8388608
  const int nw = NH_ * D_ * H_;    // 1048576
  const int nl = D_ * NH_ * H_;    // 1048576

  cvt_bf16_kernel<<<nx / 256, 256, 0, stream>>>(x, xb, nx);
  wtr_kernel<<<nw / 256, 256, 0, stream>>>(weight, wT);
  cvt_bf16_kernel<<<nl / 256, 256, 0, stream>>>(lin_w, lwb, nl);

  proj_kernel<<<128 * NH_, 128, 0, stream>>>(xb, wT, lg, lgT);

  const size_t attn_lds = 65536 + 32768 + 512 + 64 + 64;  // 98944 bytes
  attn_kernel<<<NH_ * B_ * (S_ / 16), 128, attn_lds, stream>>>(lg, lgT, attn_out, ho);

  fin_kernel<<<64 * 16, 128, 0, stream>>>(ho, lwb, lin_b, x, ytmp);

  ln_kernel<<<BS_, 256, 0, stream>>>(ytmp, ln_g, ln_b, y_out);
}
